// CrossAttention_249108103802
// MI455X (gfx1250) — matrix operand order repaired and checked by reading, not compiled, not run on hardware
//
#include <hip/hip_runtime.h>
#include <stdint.h>

// Problem constants (match reference)
#define Bn   8
#define Nn   4096
#define Cn   512
#define Hn   8
#define HDn  64
#define KVC  1024            // 2*C output columns of the kv projection
static constexpr float SCALE_ = 0.125f;   // HD^-0.5 = 64^-0.5

typedef __attribute__((ext_vector_type(16))) __bf16 v16bf;
typedef __attribute__((ext_vector_type(8)))  float  v8f;

struct __align__(16) U128 { unsigned int x, y, z, w; };
struct __align__(16) U256 { U128 lo, hi; };

static __device__ __forceinline__ v16bf mkfrag(U128 lo, U128 hi) {
  U256 t; t.lo = lo; t.hi = hi;
  return __builtin_bit_cast(v16bf, t);
}
static __device__ __forceinline__ v8f wmma_bf16(v16bf a, v16bf b, v8f c) {
  // D = A(16x32) x B(32x16) + C, f32 accumulate
  return __builtin_amdgcn_wmma_f32_16x16x32_bf16(false, a, false, b, (short)0, c, false, false);
}
// f32 -> bf16 bits, round-to-nearest-even
static __device__ __forceinline__ unsigned short f2bfu(float f) {
  unsigned int u = __builtin_bit_cast(unsigned int, f);
  u += 0x7fffu + ((u >> 16) & 1u);
  return (unsigned short)(u >> 16);
}

// ---------------------------------------------------------------------------
// Kernel 1: kv = x @ W  (f32 inputs, bf16 output), 128x128 block tile, K=512.
// grid.x = 256 * nPairs ; block = 256 (8 waves). Per wave: 32x64 tile.
#define XSTR 72   // 64 k-slice + 8 pad (multiple of 8 -> 16B aligned rows)
#define WSTR 72
__global__ __launch_bounds__(256)
void kv_gemm_kernel(const float* __restrict__ x1, const float* __restrict__ x2,
                    const float* __restrict__ W1, const float* __restrict__ W2,
                    unsigned short* __restrict__ kvws, int p0) {
  __shared__ unsigned short Xs[128 * XSTR];   // X slice  [row n][k]      bf16
  __shared__ unsigned short Ws[128 * WSTR];   // W slice  [col][k] (transposed)
  const int plocal = blockIdx.x >> 8;
  const int rem    = blockIdx.x & 255;
  const int ct = rem >> 5, nt = rem & 31;     // col-tile (8), n-tile (32)
  const int p = p0 + plocal;
  const int side = p >> 3, b = p & 7;
  const float* __restrict__ X = (side ? x2 : x1) + (size_t)b * Nn * Cn;
  const float* __restrict__ W = side ? W2 : W1;
  unsigned short* __restrict__ KV = kvws + (size_t)plocal * Nn * KVC;
  const int n0 = nt * 128, c0 = ct * 128;
  const int t = threadIdx.x;
  const int lane = t & 31, wave = t >> 5;
  const int half = lane >> 4, l15 = lane & 15;
  const int wm = wave & 3, wn = wave >> 2;    // wave tile: rows wm*32.., cols wn*64..

  v8f acc[2][4] = {};
  for (int k0 = 0; k0 < Cn; k0 += 64) {
    { // stage X tile slice [128][64] f32 -> bf16
      const int row = t >> 1, ch = t & 1;
      const float* src = X + (size_t)(n0 + row) * Cn + k0 + ch * 32;
      unsigned short* dst = &Xs[row * XSTR + ch * 32];
      #pragma unroll
      for (int j = 0; j < 8; ++j) {
        float4 f = ((const float4*)src)[j];
        dst[j*4+0] = f2bfu(f.x); dst[j*4+1] = f2bfu(f.y);
        dst[j*4+2] = f2bfu(f.z); dst[j*4+3] = f2bfu(f.w);
      }
    }
    { // stage W slice [64 k][128 c] transposed -> Ws[c][k]
      const int kk = t >> 2, ccb = (t & 3) * 32;
      const float* src = W + (size_t)(k0 + kk) * KVC + c0 + ccb;
      #pragma unroll
      for (int j = 0; j < 8; ++j) {
        float4 f = ((const float4*)src)[j];
        const int cc = ccb + j * 4;
        Ws[(cc+0)*WSTR + kk] = f2bfu(f.x);
        Ws[(cc+1)*WSTR + kk] = f2bfu(f.y);
        Ws[(cc+2)*WSTR + kk] = f2bfu(f.z);
        Ws[(cc+3)*WSTR + kk] = f2bfu(f.w);
      }
    }
    __syncthreads();
    #pragma unroll
    for (int ks = 0; ks < 64; ks += 32) {
      v16bf a[2], bb[4];
      #pragma unroll
      for (int mt = 0; mt < 2; ++mt) {
        const int row = wm*32 + mt*16 + l15;
        a[mt] = mkfrag(*(const U128*)&Xs[row*XSTR + ks + 8*half],
                       *(const U128*)&Xs[row*XSTR + ks + 16 + 8*half]);
      }
      #pragma unroll
      for (int ntt = 0; ntt < 4; ++ntt) {
        const int col = wn*64 + ntt*16 + l15;
        bb[ntt] = mkfrag(*(const U128*)&Ws[col*WSTR + ks + 8*half],
                         *(const U128*)&Ws[col*WSTR + ks + 16 + 8*half]);
      }
      #pragma unroll
      for (int mt = 0; mt < 2; ++mt)
        #pragma unroll
        for (int ntt = 0; ntt < 4; ++ntt)
          acc[mt][ntt] = wmma_bf16(a[mt], bb[ntt], acc[mt][ntt]);
    }
    __syncthreads();
  }
  #pragma unroll
  for (int mt = 0; mt < 2; ++mt)
    #pragma unroll
    for (int ntt = 0; ntt < 4; ++ntt) {
      const int gcol = c0 + wn*64 + ntt*16 + l15;
      #pragma unroll
      for (int r = 0; r < 8; ++r) {
        const int grow = n0 + wm*32 + mt*16 + 8*half + r;
        KV[(size_t)grow * KVC + gcol] = f2bfu(acc[mt][ntt][r]);
      }
    }
}

// ---------------------------------------------------------------------------
// Kernel 2: S[d][e] += sum_n K[n][d] * V[n][e]  per (pair, head).
// grid.x = 128 * nPairs ; block = 256. 16-way split over N, atomic reduce.
#define TSTR 136  // 128 n + 8 pad
__global__ __launch_bounds__(256)
void ktv_kernel(const unsigned short* __restrict__ kvws, float* __restrict__ S, int p0) {
  __shared__ unsigned short T[128 * TSTR];  // rows 0..63: K^T [d][n]; 64..127: V^T [e][n]
  const int plocal = blockIdx.x >> 7;
  const int rem = blockIdx.x & 127;
  const int h = rem >> 4, split = rem & 15;
  const int p = p0 + plocal;
  const unsigned short* __restrict__ KV = kvws + (size_t)plocal * Nn * KVC;
  float* __restrict__ Sb = S + ((size_t)p * Hn + h) * HDn * HDn;
  const int t = threadIdx.x;
  const int lane = t & 31, wave = t >> 5;
  const int half = lane >> 4, l15 = lane & 15;
  const int dt = wave & 3, eg = wave >> 2;  // wave: d-tile dt, e-tiles eg*2, eg*2+1
  const int c = t & 127, rr0 = (t >> 7) * 64;
  const int gcol = (c < 64) ? (h*HDn + c) : (Cn + h*HDn + (c - 64));
  v8f acc[2] = {};
  for (int sub = 0; sub < 2; ++sub) {
    const int nb = split * 256 + sub * 128;
    for (int rr = 0; rr < 64; ++rr) {       // transpose-stage 128 rows of k|v
      const int r = rr0 + rr;
      T[c * TSTR + r] = KV[(size_t)(nb + r) * KVC + gcol];
    }
    __syncthreads();
    #pragma unroll
    for (int n0 = 0; n0 < 128; n0 += 32) {
      const int drow = dt*16 + l15;
      v16bf a = mkfrag(*(const U128*)&T[drow*TSTR + n0 + 8*half],
                       *(const U128*)&T[drow*TSTR + n0 + 16 + 8*half]);
      #pragma unroll
      for (int j = 0; j < 2; ++j) {
        const int erow = 64 + eg*32 + j*16 + l15;
        v16bf bb = mkfrag(*(const U128*)&T[erow*TSTR + n0 + 8*half],
                          *(const U128*)&T[erow*TSTR + n0 + 16 + 8*half]);
        acc[j] = wmma_bf16(a, bb, acc[j]);
      }
    }
    __syncthreads();
  }
  #pragma unroll
  for (int j = 0; j < 2; ++j) {
    const int e = eg*32 + j*16 + l15;
    #pragma unroll
    for (int r = 0; r < 8; ++r) {
      const int d = dt*16 + 8*half + r;
      atomicAdd(&Sb[d*HDn + e], acc[j][r]);
    }
  }
}

// ---------------------------------------------------------------------------
// Kernel 3: ctx = softmax(S * SCALE, over d axis); write ctx^T [e][d] in bf16.
__global__ __launch_bounds__(64)
void softmax_kernel(const float* __restrict__ S, unsigned short* __restrict__ CT) {
  const int blk = blockIdx.x;   // p*8 + h, 0..127
  const int e = threadIdx.x;    // 0..63
  const float* Sb = S + (size_t)blk * (HDn*HDn);
  unsigned short* Cb = CT + (size_t)blk * (HDn*HDn);
  float mx = -3.402823e38f;
  for (int d = 0; d < 64; ++d) mx = fmaxf(mx, Sb[d*64 + e] * SCALE_);
  float sum = 0.f;
  for (int d = 0; d < 64; ++d) sum += __expf(Sb[d*64 + e] * SCALE_ - mx);
  const float inv = 1.f / sum;
  for (int d = 0; d < 64; ++d)
    Cb[e*64 + d] = f2bfu(__expf(Sb[d*64 + e] * SCALE_ - mx) * inv);
}

// ---------------------------------------------------------------------------
// Kernel 4: o_s[b,n,h*64+e] = sum_d x_s[b,n,h*64+d] * ctx_{1-s}[b,h,d,e]
// grid.x = 4096 (2 sides * 8 b * 8 h * 32 n-tiles); block = 256.
__global__ __launch_bounds__(256)
void out_gemm_kernel(const float* __restrict__ x1, const float* __restrict__ x2,
                     const unsigned short* __restrict__ CT, float* __restrict__ out) {
  __shared__ unsigned short Xs[128 * XSTR];
  __shared__ unsigned short Cs[64 * 64];   // ctx^T [e][d]
  const int id = blockIdx.x;
  const int nt = id & 31, h = (id >> 5) & 7, b = (id >> 8) & 7, side = id >> 11;
  const float* __restrict__ X = (side ? x2 : x1) + (size_t)b * Nn * Cn;
  const unsigned short* __restrict__ Csrc =
      CT + ((size_t)((1 - side) * 8 + b) * Hn + h) * (HDn*HDn);  // CROSS pairing
  const int n0 = nt * 128;
  const int t = threadIdx.x;
  const int lane = t & 31, wave = t >> 5;
  const int half = lane >> 4, l15 = lane & 15;
  { // stage q tile [128][64] f32 -> bf16
    const int row = t >> 1, ch = t & 1;
    const float* src = X + (size_t)(n0 + row) * Cn + h * HDn + ch * 32;
    unsigned short* dst = &Xs[row * XSTR + ch * 32];
    #pragma unroll
    for (int j = 0; j < 8; ++j) {
      float4 f = ((const float4*)src)[j];
      dst[j*4+0] = f2bfu(f.x); dst[j*4+1] = f2bfu(f.y);
      dst[j*4+2] = f2bfu(f.z); dst[j*4+3] = f2bfu(f.w);
    }
  }
  #pragma unroll
  for (int i = 0; i < 2; ++i)   // stage ctx^T (8 KB)
    ((U128*)Cs)[t*2 + i] = ((const U128*)Csrc)[t*2 + i];
  __syncthreads();

  v8f acc[4] = {};
  #pragma unroll
  for (int ks = 0; ks < 64; ks += 32) {
    const int row = wave*16 + l15;
    v16bf a = mkfrag(*(const U128*)&Xs[row*XSTR + ks + 8*half],
                     *(const U128*)&Xs[row*XSTR + ks + 16 + 8*half]);
    #pragma unroll
    for (int et = 0; et < 4; ++et) {
      const int e = et*16 + l15;
      v16bf bb = mkfrag(*(const U128*)&Cs[e*64 + ks + 8*half],
                        *(const U128*)&Cs[e*64 + ks + 16 + 8*half]);
      acc[et] = wmma_bf16(a, bb, acc[et]);
    }
  }
  float* __restrict__ O = out + (size_t)side * Bn * Nn * Cn + (size_t)b * Nn * Cn;
  #pragma unroll
  for (int et = 0; et < 4; ++et) {
    const int e = h*HDn + et*16 + l15;
    #pragma unroll
    for (int r = 0; r < 8; ++r) {
      const int n = n0 + wave*16 + 8*half + r;
      O[(size_t)n * Cn + e] = acc[et][r];
    }
  }
}

// ---------------------------------------------------------------------------
extern "C" void kernel_launch(void* const* d_in, const int* in_sizes, int n_in,
                              void* d_out, int out_size, void* d_ws, size_t ws_size,
                              hipStream_t stream) {
  (void)in_sizes; (void)n_in; (void)out_size;
  const float* x1 = (const float*)d_in[0];
  const float* x2 = (const float*)d_in[1];
  const float* W1 = (const float*)d_in[2];
  const float* W2 = (const float*)d_in[3];
  float* out = (float*)d_out;
  char* ws = (char*)d_ws;

  // Workspace layout: S f32 (2 MB) | ctx^T bf16 (1 MB) | pad | kv bf16 chunks
  float*          S  = (float*)ws;
  unsigned short* CT = (unsigned short*)(ws + ((size_t)2 << 20));
  unsigned short* kv = (unsigned short*)(ws + ((size_t)4 << 20));
  const size_t kv_bytes_per_p = (size_t)Nn * KVC * sizeof(unsigned short); // 8 MB
  size_t avail = (ws_size > ((size_t)4 << 20)) ? ws_size - ((size_t)4 << 20) : kv_bytes_per_p;
  int chunk = (int)(avail / kv_bytes_per_p);
  if (chunk < 1)  chunk = 1;
  if (chunk > 16) chunk = 16;

  hipMemsetAsync(S, 0, (size_t)16 * Hn * HDn * HDn * sizeof(float), stream);
  for (int p0 = 0; p0 < 16; p0 += chunk) {
    const int c = (16 - p0 < chunk) ? (16 - p0) : chunk;
    kv_gemm_kernel<<<dim3(256u * c), dim3(256), 0, stream>>>(x1, x2, W1, W2, kv, p0);
    ktv_kernel   <<<dim3(128u * c), dim3(256), 0, stream>>>(kv, S, p0);
  }
  softmax_kernel <<<dim3(128),  dim3(64),  0, stream>>>(S, CT);
  out_gemm_kernel<<<dim3(4096), dim3(256), 0, stream>>>(x1, x2, CT, out);
}
